// LSTMAutoencoder_25975962206284
// MI455X (gfx1250) — matrix operand order repaired and checked by reading, not compiled, not run on hardware
//
#include <hip/hip_runtime.h>
#include <math.h>

typedef _Float16 v16h __attribute__((ext_vector_type(16)));
typedef _Float16 v8h  __attribute__((ext_vector_type(8)));
typedef _Float16 v4h  __attribute__((ext_vector_type(4)));
typedef float    v8f  __attribute__((ext_vector_type(8)));
typedef float    v4f  __attribute__((ext_vector_type(4)));

constexpr int kB = 512;   // batch
constexpr int kT = 256;   // seq len
constexpr int kD = 64;    // input dim
constexpr int kH = 64;    // hidden dim
constexpr int kG = 256;   // 4*H gate width

// ---------------------------------------------------------------------------
// WMMA helpers (CDNA5 / gfx1250, wave32)
// ---------------------------------------------------------------------------
__device__ __forceinline__ v8f wmma_f16(v16h a, v16h b, v8f c) {
  // v_wmma_f32_16x16x32_f16  D = A(16x32) * B(32x16) + C(16x16 f32)
  return __builtin_amdgcn_wmma_f32_16x16x32_f16(
      /*neg_a=*/false, a, /*neg_b=*/false, b,
      /*c_mod=*/(short)0, c, /*reuse_a=*/false, /*reuse_b=*/false);
}

// A-fragment (16x32 f16, MxK) from row-major f16 [rows][ldk].
// Lane L: m = L%16 ; elements e<8 -> K = koff+sel+e ; e>=8 -> K = koff+sel+16+(e-8)
// with sel = 8*(L>=16).  Two 16-byte contiguous loads.
__device__ __forceinline__ v16h load_afrag_f16(const _Float16* base, int lane,
                                               int koff, int ldk) {
  int m   = lane & 15;
  int sel = (lane >> 4) * 8;
  const _Float16* p = base + (size_t)m * ldk + koff + sel;
  v8h lo = *(const v8h*)(p);
  v8h hi = *(const v8h*)(p + 16);
  return __builtin_shufflevector(lo, hi, 0, 1, 2, 3, 4, 5, 6, 7,
                                 8, 9, 10, 11, 12, 13, 14, 15);
}

// A-fragment from f32 input x laid out [B][T][D]; logical GEMM row r = t*B + b.
__device__ __forceinline__ v16h load_afrag_f32map(const float* x, int r0,
                                                  int lane, int koff) {
  int m   = lane & 15;
  int sel = (lane >> 4) * 8;
  int r   = r0 + m;
  int t   = r >> 9;    // / kB (512)
  int b   = r & 511;
  const float* p = x + ((size_t)b * kT + t) * kD + koff + sel;
  v4f f0 = *(const v4f*)(p);
  v4f f1 = *(const v4f*)(p + 4);
  v4f f2 = *(const v4f*)(p + 16);
  v4f f3 = *(const v4f*)(p + 20);
  v16h a;
#pragma unroll
  for (int i = 0; i < 4; ++i) {
    a[i]      = (_Float16)f0[i];
    a[4 + i]  = (_Float16)f1[i];
    a[8 + i]  = (_Float16)f2[i];
    a[12 + i] = (_Float16)f3[i];
  }
  return a;
}

// B-fragment (32x16 f16, KxN) for D[m][n] = sum_k A[m][k] * W[n][k]:
// B[k][n] = W[n][k], W row-major [N][ldk].  Lane L: n = L%16,
// elements e<8: K = koff + 8*(L>=16) + e ; e>=8: K = koff + 8*(L>=16) + 16 + (e-8).
__device__ __forceinline__ v16h load_bfrag(const _Float16* W, int n0, int koff,
                                           int ldk, int lane) {
  int n    = lane & 15;
  int ksel = (lane >> 4) * 8;
  const _Float16* p = W + (size_t)(n0 + n) * ldk + koff + ksel;
  v8h lo = *(const v8h*)(p);
  v8h hi = *(const v8h*)(p + 16);
  return __builtin_shufflevector(lo, hi, 0, 1, 2, 3, 4, 5, 6, 7,
                                 8, 9, 10, 11, 12, 13, 14, 15);
}

// Async global -> LDS copy of 16 bytes (CDNA5, tracked by ASYNCcnt).
// lds_off: byte offset in LDS (low 32 bits of generic shared pointer).
__device__ __forceinline__ void async_copy_b128(unsigned lds_off, const void* g) {
  asm volatile("global_load_async_to_lds_b128 %0, %1, off"
               :: "v"(lds_off), "v"((unsigned long long)(uintptr_t)g)
               : "memory");
}

__device__ __forceinline__ float fsig(float x) {
  return 1.0f / (1.0f + __expf(-x));
}
__device__ __forceinline__ float ftanh(float x) {
  return 2.0f * fsig(2.0f * x) - 1.0f;
}

// ---------------------------------------------------------------------------
// Weight prep: f32 -> f16 weights, combined bias
// ---------------------------------------------------------------------------
__global__ void prep_layer(const float* __restrict__ Wih,
                           const float* __restrict__ Whh,
                           const float* __restrict__ bih,
                           const float* __restrict__ bhh,
                           _Float16* __restrict__ Wih_h,
                           _Float16* __restrict__ Whh_h,
                           float* __restrict__ bias) {
  int i = blockIdx.x * blockDim.x + threadIdx.x;
  if (i < kG * kH) {
    Wih_h[i] = (_Float16)Wih[i];
    Whh_h[i] = (_Float16)Whh[i];
  }
  if (i < kG) bias[i] = bih[i] + bhh[i];
}

__global__ void prep_fc(const float* __restrict__ W, const float* __restrict__ b,
                        _Float16* __restrict__ Wh, float* __restrict__ bf) {
  int i = blockIdx.x * blockDim.x + threadIdx.x;
  if (i < kD * kH) Wh[i] = (_Float16)W[i];
  if (i < kD) bf[i] = b[i];
}

// ---------------------------------------------------------------------------
// xg GEMM:  out[r][n] = sum_k in[r][k] * W[n][k] + bias[n]   (N = 256, K = 64)
// out is f16 [R][256] with r = t*B + b (time-major rows).
// Block: 256 threads = 8 waves; tile 32 rows x 256 cols (wave = 16x64).
// ---------------------------------------------------------------------------
template <bool IN_F32_BTD>
__global__ void gemm_xg(const void* __restrict__ in_,
                        const _Float16* __restrict__ W,
                        const float* __restrict__ bias,
                        _Float16* __restrict__ out) {
  int tid  = threadIdx.x;
  int lane = tid & 31;
  int w    = tid >> 5;
  int r0   = blockIdx.x * 32 + (w >> 2) * 16;
  int n0   = (w & 3) * 64;

  v16h a0, a1;
  if (IN_F32_BTD) {
    const float* x = (const float*)in_;
    a0 = load_afrag_f32map(x, r0, lane, 0);
    a1 = load_afrag_f32map(x, r0, lane, 32);
  } else {
    const _Float16* xh = (const _Float16*)in_ + (size_t)r0 * kH;
    a0 = load_afrag_f16(xh, lane, 0, kH);
    a1 = load_afrag_f16(xh, lane, 32, kH);
  }

  int mo = (lane >> 4) * 8;
  int nl = lane & 15;
#pragma unroll
  for (int nt = 0; nt < 4; ++nt) {
    int n = n0 + nt * 16;
    v16h b0 = load_bfrag(W, n, 0, kH, lane);
    v16h b1 = load_bfrag(W, n, 32, kH, lane);
    v8f acc = {};
    acc = wmma_f16(a0, b0, acc);
    acc = wmma_f16(a1, b1, acc);
    float bv = bias[n + nl];
    _Float16* op = out + (size_t)r0 * kG + n + nl;
#pragma unroll
    for (int r = 0; r < 8; ++r) {
      op[(size_t)(r + mo) * kG] = (_Float16)(acc[r] + bv);
    }
  }
}

// ---------------------------------------------------------------------------
// LSTM recurrence over T steps for a 16-row batch tile.
// xg: [TSTRIDE? T : 1][B][256] f16 (bias folded in); Whh: [256][64] f16;
// hseq out: [T][B][64] f16.  256 threads / 8 waves; wave w owns gate columns
// [32w, 32w+32); Whh B-fragments preloaded (loop invariant).
// h lives in LDS (f16), c in registers (f32), gate exchange via LDS f32.
// xg tiles are double-buffered in LDS via async global->LDS loads (ASYNCcnt),
// prefetched one timestep ahead so global latency is off the serial path.
// ---------------------------------------------------------------------------
template <int TSTRIDE>
__global__ void lstm_rec(const _Float16* __restrict__ xg,
                         const _Float16* __restrict__ Whh,
                         _Float16* __restrict__ hseq) {
  __shared__ __align__(32) _Float16 hbuf[16 * 64];      // h_{t-1}, A-frag source
  __shared__ __align__(16) float    gbuf[16 * 256];     // pre-activation gates
  __shared__ __align__(16) _Float16 xbuf[2][16 * 256];  // staged xg tiles (2x8KB)

  int tid  = threadIdx.x;
  int lane = tid & 31;
  int w    = tid >> 5;
  int b0   = blockIdx.x * 16;

  // Loop-invariant Whh fragments for this wave's two 16-col N tiles.
  int na = 32 * w;
  int nb = na + 16;
  v16h bA0 = load_bfrag(Whh, na, 0, kH, lane);
  v16h bA1 = load_bfrag(Whh, na, 32, kH, lane);
  v16h bB0 = load_bfrag(Whh, nb, 0, kH, lane);
  v16h bB1 = load_bfrag(Whh, nb, 32, kH, lane);

  // Async prefetch bookkeeping: this block's xg tile (16 rows) is one
  // contiguous 8KB chunk in the [T][B][256] layout. Each thread moves 2x16B.
  unsigned x0off = (unsigned)(uintptr_t)(&xbuf[0][0]);
  unsigned x1off = (unsigned)(uintptr_t)(&xbuf[1][0]);
  unsigned co0   = (unsigned)tid * 16u;
  unsigned co1   = 4096u + (unsigned)tid * 16u;
  const char* gtile0 = (const char*)(xg + (size_t)b0 * kG);
  const size_t tbytes = (size_t)kB * kG * sizeof(_Float16);

  // Prefetch tile t=0.
  async_copy_b128(x0off + co0, gtile0 + co0);
  async_copy_b128(x0off + co1, gtile0 + co1);

  // h0 = 0
  ((float*)hbuf)[tid]       = 0.0f;
  ((float*)hbuf)[tid + 256] = 0.0f;

  // Elementwise ownership: thread -> (row, j0..j0+3), c state in registers.
  int row = tid >> 4;
  int j0  = (tid & 15) * 4;
  float cst[4] = {0.0f, 0.0f, 0.0f, 0.0f};

  int mo = (lane >> 4) * 8;
  int nl = lane & 15;
  __syncthreads();

  for (int t = 0; t < kT; ++t) {
    // g = h_{t-1} @ Whh^T  (16x64 * 64x32 per wave)
    v16h a0 = load_afrag_f16(hbuf, lane, 0, kH);
    v16h a1 = load_afrag_f16(hbuf, lane, 32, kH);
    v8f accA = {}, accB = {};
    accA = wmma_f16(a0, bA0, accA);
    accA = wmma_f16(a1, bA1, accA);
    accB = wmma_f16(a0, bB0, accB);
    accB = wmma_f16(a1, bB1, accB);
#pragma unroll
    for (int r = 0; r < 8; ++r) {
      gbuf[(r + mo) * kG + na + nl] = accA[r];
      gbuf[(r + mo) * kG + nb + nl] = accB[r];
    }

    if (TSTRIDE) {
      // Prefetch tile t+1 into the other buffer (clamped re-fetch at the end
      // keeps the outstanding-op count uniform so the wait below is exact).
      int tn = (t + 1 < kT) ? (t + 1) : t;
      const char* gp = gtile0 + (size_t)tn * tbytes;
      unsigned xo = ((t + 1) & 1) ? x1off : x0off;
      async_copy_b128(xo + co0, gp + co0);
      async_copy_b128(xo + co1, gp + co1);
      // Async loads complete in order: <=2 outstanding means tile t is ready.
      asm volatile("s_wait_asynccnt 0x2" ::: "memory");
    } else {
      asm volatile("s_wait_asynccnt 0x0" ::: "memory");
    }
    __syncthreads();

    // Gates + state update for 4 hidden units of one batch row (xg from LDS).
    const _Float16* xr = &xbuf[TSTRIDE ? (t & 1) : 0][row * kG + j0];
    const float* grow  = gbuf + row * kG + j0;
    v4h hv;
#pragma unroll
    for (int q = 0; q < 4; ++q) {
      float ip = grow[q]       + (float)xr[q];
      float fp = grow[64 + q]  + (float)xr[64 + q];
      float gp = grow[128 + q] + (float)xr[128 + q];
      float op = grow[192 + q] + (float)xr[192 + q];
      float ig = fsig(ip);
      float fg = fsig(fp);
      float gg = ftanh(gp);
      float og = fsig(op);
      cst[q] = fg * cst[q] + ig * gg;
      hv[q]  = (_Float16)(og * ftanh(cst[q]));
    }
    *(v4h*)(hbuf + row * kH + j0) = hv;
    *(v4h*)(hseq + ((size_t)t * kB + b0 + row) * kH + j0) = hv;
    __syncthreads();
  }
}

// ---------------------------------------------------------------------------
// Final FC: out[b][t][d] = sum_j dseq[t*B+b][j] * fcW[d][j] + fcb[d]  (f32 out)
// Block: 8 waves, each 16 rows x 64 cols -> 128 rows/block.
// ---------------------------------------------------------------------------
__global__ void gemm_fc(const _Float16* __restrict__ in,
                        const _Float16* __restrict__ W,
                        const float* __restrict__ bias,
                        float* __restrict__ out) {
  int tid  = threadIdx.x;
  int lane = tid & 31;
  int w    = tid >> 5;
  int r0   = blockIdx.x * 128 + w * 16;

  const _Float16* xh = in + (size_t)r0 * kH;
  v16h a0 = load_afrag_f16(xh, lane, 0, kH);
  v16h a1 = load_afrag_f16(xh, lane, 32, kH);

  int mo = (lane >> 4) * 8;
  int nl = lane & 15;
#pragma unroll
  for (int nt = 0; nt < 4; ++nt) {
    int n = nt * 16;
    v16h b0 = load_bfrag(W, n, 0, kH, lane);
    v16h b1 = load_bfrag(W, n, 32, kH, lane);
    v8f acc = {};
    acc = wmma_f16(a0, b0, acc);
    acc = wmma_f16(a1, b1, acc);
    float bv = bias[n + nl];
#pragma unroll
    for (int r = 0; r < 8; ++r) {
      int gr = r0 + r + mo;        // = t*B + b
      int t  = gr >> 9;
      int b  = gr & 511;
      out[((size_t)b * kT + t) * kD + n + nl] = acc[r] + bv;
    }
  }
}

// ---------------------------------------------------------------------------
// Host-side orchestration
// ---------------------------------------------------------------------------
extern "C" void kernel_launch(void* const* d_in, const int* in_sizes, int n_in,
                              void* d_out, int out_size, void* d_ws,
                              size_t ws_size, hipStream_t stream) {
  (void)in_sizes; (void)n_in; (void)out_size; (void)ws_size;

  const float* x = (const float*)d_in[0];
  const float* Wih[4] = {(const float*)d_in[1], (const float*)d_in[5],
                         (const float*)d_in[9], (const float*)d_in[13]};
  const float* Whh[4] = {(const float*)d_in[2], (const float*)d_in[6],
                         (const float*)d_in[10], (const float*)d_in[14]};
  const float* bih[4] = {(const float*)d_in[3], (const float*)d_in[7],
                         (const float*)d_in[11], (const float*)d_in[15]};
  const float* bhh[4] = {(const float*)d_in[4], (const float*)d_in[8],
                         (const float*)d_in[12], (const float*)d_in[16]};
  const float* fcW = (const float*)d_in[17];
  const float* fcb = (const float*)d_in[18];

  char*  ws  = (char*)d_ws;
  size_t off = 0;
  auto alloc = [&](size_t bytes) -> void* {
    off = (off + 255) & ~(size_t)255;
    void* p = ws + off;
    off += bytes;
    return p;
  };

  _Float16* Wih_h[4];
  _Float16* Whh_h[4];
  float*    bias[4];
  for (int l = 0; l < 4; ++l) {
    Wih_h[l] = (_Float16*)alloc((size_t)kG * kH * 2);
    Whh_h[l] = (_Float16*)alloc((size_t)kG * kH * 2);
    bias[l]  = (float*)alloc((size_t)kG * 4);
  }
  _Float16* fcW_h = (_Float16*)alloc((size_t)kD * kH * 2);
  float*    fcb_f = (float*)alloc((size_t)kD * 4);
  _Float16* xgbuf = (_Float16*)alloc((size_t)kT * kB * kG * 2);  // 64 MB
  _Float16* hA    = (_Float16*)alloc((size_t)kT * kB * kH * 2);  // 16 MB
  _Float16* hB    = (_Float16*)alloc((size_t)kT * kB * kH * 2);  // 16 MB

  // Weight conversion (f32 -> f16, combined bias).
  for (int l = 0; l < 4; ++l)
    prep_layer<<<64, 256, 0, stream>>>(Wih[l], Whh[l], bih[l], bhh[l],
                                       Wih_h[l], Whh_h[l], bias[l]);
  prep_fc<<<16, 256, 0, stream>>>(fcW, fcb, fcW_h, fcb_f);

  const int gemmBlocks = (kT * kB) / 32;  // 4096
  const int recBlocks  = kB / 16;         // 32

  // Encoder layer 0: x [B,T,D] f32 -> hA
  gemm_xg<true><<<gemmBlocks, 256, 0, stream>>>(x, Wih_h[0], bias[0], xgbuf);
  lstm_rec<1><<<recBlocks, 256, 0, stream>>>(xgbuf, Whh_h[0], hA);

  // Encoder layer 1: hA -> hB (latent = hB[T-1])
  gemm_xg<false><<<gemmBlocks, 256, 0, stream>>>(hA, Wih_h[1], bias[1], xgbuf);
  lstm_rec<1><<<recBlocks, 256, 0, stream>>>(xgbuf, Whh_h[1], hB);

  // Decoder layer 0: latent repeated across time -> xg is t-invariant ([B,256]).
  const _Float16* latent = hB + (size_t)(kT - 1) * kB * kH;
  gemm_xg<false><<<kB / 32, 256, 0, stream>>>(latent, Wih_h[2], bias[2], xgbuf);
  lstm_rec<0><<<recBlocks, 256, 0, stream>>>(xgbuf, Whh_h[2], hA);

  // Decoder layer 1: hA -> hB
  gemm_xg<false><<<gemmBlocks, 256, 0, stream>>>(hA, Wih_h[3], bias[3], xgbuf);
  lstm_rec<1><<<recBlocks, 256, 0, stream>>>(xgbuf, Whh_h[3], hB);

  // Output projection -> d_out [B,T,D] f32
  gemm_fc<<<(kT * kB) / 128, 256, 0, stream>>>(hB, fcW_h, fcb_f, (float*)d_out);
}
